// GCNTransformerEncoder_73418170958475
// MI455X (gfx1250) — matrix operand order repaired and checked by reading, not compiled, not run on hardware
//
#include <hip/hip_runtime.h>

typedef __attribute__((ext_vector_type(16))) _Float16 v16h;
typedef __attribute__((ext_vector_type(8)))  float    v8f;

// Problem dims
constexpr int Bq  = 64;     // batch
constexpr int Nq  = 256;    // nodes
constexpr int INq = 128;    // input feat
constexpr int Eq  = 256;    // embed
constexpr int Hq  = 8;      // heads
constexpr int Fq  = 1024;   // ffn

// ---------------------------------------------------------------------------
// wave32 reductions
// ---------------------------------------------------------------------------
__device__ __forceinline__ float wave_sum(float v) {
#pragma unroll
  for (int o = 16; o > 0; o >>= 1) v += __shfl_xor(v, o, 32);
  return v;
}
__device__ __forceinline__ float wave_max(float v) {
#pragma unroll
  for (int o = 16; o > 0; o >>= 1) v = fmaxf(v, __shfl_xor(v, o, 32));
  return v;
}

// ---------------------------------------------------------------------------
// Adjacency normalization
// ---------------------------------------------------------------------------
__global__ __launch_bounds__(256) void k_rowdis(const float* __restrict__ adj,
                                                float* __restrict__ dis) {
  const int wave = threadIdx.x >> 5, lane = threadIdx.x & 31;
  const int row = blockIdx.x * 8 + wave;      // [0, B*N)
  const int i = row & (Nq - 1);
  const float* ar = adj + (size_t)row * Nq;
  float s = 0.f;
#pragma unroll
  for (int j = 0; j < 8; ++j) {
    const int c = lane * 8 + j;
    float a = ar[c];
    if (c == i) a = 1.f;
    s += a;
  }
  s = wave_sum(s);
  if (lane == 0) dis[row] = rsqrtf(fmaxf(s, 1.f));
}

__global__ __launch_bounds__(256) void k_An(const float* __restrict__ adj,
                                            const float* __restrict__ dis,
                                            float* __restrict__ An) {
  const size_t idx = (size_t)blockIdx.x * 256 + threadIdx.x;   // [0, B*N*N)
  const int j = (int)(idx & (Nq - 1));
  const size_t r = idx >> 8;
  const int i = (int)(r & (Nq - 1));
  const int b = (int)(r >> 8);
  const float a = (i == j) ? 1.f : adj[idx];
  An[idx] = dis[(b << 8) + i] * a * dis[(b << 8) + j];
}

// ---------------------------------------------------------------------------
// Row layernorm over E=256 (one wave per row)
// ---------------------------------------------------------------------------
__global__ __launch_bounds__(256) void k_ln(const float* __restrict__ x,
                                            const float* __restrict__ g,
                                            const float* __restrict__ b,
                                            float* __restrict__ y) {
  const int wave = threadIdx.x >> 5, lane = threadIdx.x & 31;
  const int row = blockIdx.x * 8 + wave;      // [0, B*N)
  const float* xr = x + (size_t)row * Eq;
  float v[8];
#pragma unroll
  for (int i = 0; i < 8; ++i) v[i] = xr[lane * 8 + i];
  float s = 0.f;
#pragma unroll
  for (int i = 0; i < 8; ++i) s += v[i];
  const float m = wave_sum(s) * (1.f / Eq);
  float q = 0.f;
#pragma unroll
  for (int i = 0; i < 8; ++i) { const float d = v[i] - m; q += d * d; }
  const float inv = rsqrtf(wave_sum(q) * (1.f / Eq) + 1e-5f);
  float* yr = y + (size_t)row * Eq;
#pragma unroll
  for (int i = 0; i < 8; ++i) {
    const int c = lane * 8 + i;
    yr[c] = (v[i] - m) * inv * g[c] + b[c];
  }
}

// ---------------------------------------------------------------------------
// Masked softmax: S (f32, B*H*N rows of 256) -> P (f16)
// ---------------------------------------------------------------------------
__global__ __launch_bounds__(256) void k_softmax(const float* __restrict__ S,
                                                 const float* __restrict__ mask,
                                                 _Float16* __restrict__ P) {
  const int wave = threadIdx.x >> 5, lane = threadIdx.x & 31;
  const long rid = (long)blockIdx.x * 8 + wave;       // [0, B*H*N)
  const int zz = (int)(rid >> 8);                     // b*H + h
  const int n = (int)(rid & (Nq - 1));
  const int b = zz >> 3;
  const float* sr = S + rid * Nq;
  float v[8];
#pragma unroll
  for (int i = 0; i < 8; ++i) v[i] = sr[lane * 8 + i];
  float mx = -1e30f;
#pragma unroll
  for (int i = 0; i < 8; ++i) mx = fmaxf(mx, v[i]);
  mx = wave_max(mx);
  float sum = 0.f;
#pragma unroll
  for (int i = 0; i < 8; ++i) { v[i] = __expf(v[i] - mx); sum += v[i]; }
  sum = wave_sum(sum);
  const float inv = 1.f / sum;
  const float mr = mask[(b << 8) + n];
  _Float16* pr = P + rid * Nq;
#pragma unroll
  for (int i = 0; i < 8; ++i) {
    const int c = lane * 8 + i;
    const float am = mr * mask[(b << 8) + c];
    pr[c] = (_Float16)((am == 0.f) ? 0.f : v[i] * inv);
  }
}

// ---------------------------------------------------------------------------
// Fused epilogues
// ---------------------------------------------------------------------------
struct EpiP {
  float* out_f;
  _Float16* out_h;
  const float* bias;
  const float* mask;   // (B,N) flat
  const float* dist;   // (B,N,N)
  const float* add1;   // gcn_res
  const float* add2;   // att
  float* accum;        // gcn accumulation
  float scale;
  int first;
  int ldc;
};

template <int MODE>
__device__ __forceinline__ void epi_store(const EpiP& p, int z, int row, int col, float v) {
  if constexpr (MODE == 0) {            // plain f32 store
    p.out_f[(size_t)row * p.ldc + col] = v;
  } else if constexpr (MODE == 1) {     // GCN: relu(acc+bias)*mask, accumulate gcn_res
    const size_t idx = ((size_t)z << 16) + ((size_t)row << 8) + col;
    const float g = fmaxf(v + p.bias[col], 0.f) * p.mask[(z << 8) + row];
    p.out_f[idx] = g;
    if (p.first) p.accum[idx] = g; else p.accum[idx] += g;
  } else if constexpr (MODE == 2) {     // Q/K: scale*mask, (B,H,N,E) f16
    const int b = row >> 8, n = row & 255, hh = col >> 8, e = col & 255;
    const float val = v * p.scale * p.mask[row];
    const size_t idx = ((((size_t)((b << 3) + hh)) << 8) + n) * Eq + e;
    p.out_h[idx] = (_Float16)val;
  } else if constexpr (MODE == 3) {     // scores: +dist, mask -> -1e9
    const int b = z >> 3;
    const float am = p.mask[(b << 8) + row] * p.mask[(b << 8) + col];
    float val = v + p.dist[(((size_t)b << 8) + row) * Nq + col];
    if (am == 0.f) val = -1e9f;
    p.out_f[((size_t)z << 16) + ((size_t)row << 8) + col] = val;
  } else if constexpr (MODE == 4) {     // y: (B,N,H*E) f16 store
    const int b = z >> 3, hh = z & 7;
    const size_t idx = (((size_t)(b << 8) + row) << 11) + (hh << 8) + col;
    p.out_h[idx] = (_Float16)v;
  } else if constexpr (MODE == 5) {     // Wo: att = gcn + (acc+bo)*mask
    const size_t idx = ((size_t)row << 8) + col;
    const float val = (v + p.bias[col]) * p.mask[row];
    p.out_f[idx] = p.add1[idx] + val;
  } else if constexpr (MODE == 6) {     // FFN1: exact gelu, f16 store
    const float x = v + p.bias[col];
    const float g = 0.5f * x * (1.f + erff(x * 0.70710678118f));
    p.out_h[((size_t)row << 10) + col] = (_Float16)g;
  } else if constexpr (MODE == 7) {     // out = gcn + att + (acc+bf2)*mask
    const size_t idx = ((size_t)row << 8) + col;
    const float f = (v + p.bias[col]) * p.mask[row];
    p.out_f[idx] = p.add1[idx] + p.add2[idx] + f;
  } else {                              // MODE 8: V transposed: (B,H,E,N) f16, masked
    const int b = row >> 8, n = row & 255, hh = col >> 8, e = col & 255;
    const float val = v * p.mask[row];
    const size_t idx = ((((size_t)((b << 3) + hh)) << 8) + e) * Nq + n;
    p.out_h[idx] = (_Float16)val;
  }
}

// ---------------------------------------------------------------------------
// Tiled WMMA GEMM: C(MxN) = A(MxK)*B(KxN), f16 compute, f32 accumulate.
//  AH/BH: global operand already f16.  BT: B accessed k-contiguous (B[k][n]=Bg[n*ldb+k]).
//  Block 128x128x32, 8 waves, wave tile 64x32 = 4x2 v_wmma_f32_16x16x32_f16.
//  B staged TRANSPOSED in LDS (BsT[n][k]) so both A and B fragments are
//  2x contiguous ds_load_b128.  All-f16 staging uses global_load_async_to_lds_b128.
// ---------------------------------------------------------------------------
template <int MODE, bool AH, bool BH, bool BT>
__global__ __launch_bounds__(256) void gemm_wmma_k(
    const void* __restrict__ Ag, const void* __restrict__ Bg,
    int Kdim, int lda, int ldb, long aBatch, long bBatch, EpiP p) {
  constexpr int BM = 128, BN = 128, BK = 32, PAD = 8, LDK = BK + PAD;  // 40 halfs = 80 B
  __shared__ __align__(16) _Float16 As[BM * LDK];
  __shared__ __align__(16) _Float16 BsT[BN * LDK];

  const int tid = threadIdx.x, lane = tid & 31, wave = tid >> 5;
  const int wm = wave >> 2, wn = wave & 3;
  const int z = blockIdx.z;
  const int rowBase = blockIdx.y * BM, colBase = blockIdx.x * BN;

  const _Float16* A16 = (const _Float16*)Ag + (AH ? (size_t)z * aBatch : 0);
  const float*    A32 = (const float*)Ag + (AH ? 0 : (size_t)z * aBatch);
  const _Float16* B16 = (const _Float16*)Bg + (BH ? (size_t)z * bBatch : 0);
  const float*    B32 = (const float*)Bg + (BH ? 0 : (size_t)z * bBatch);

  const v8f vzero = {0.f, 0.f, 0.f, 0.f, 0.f, 0.f, 0.f, 0.f};
  v8f acc[4][2];
#pragma unroll
  for (int i = 0; i < 4; ++i)
#pragma unroll
    for (int j = 0; j < 2; ++j) acc[i][j] = vzero;

  const int nkt = Kdim / BK;
  for (int kt = 0; kt < nkt; ++kt) {
    const int k0 = kt * BK;

    // ---------------- stage A tile (BM x BK) ----------------
    {
      const int r = tid >> 1, ks = (tid & 1) * 16;
      _Float16* dst = &As[r * LDK + ks];
      if constexpr (AH) {
        const _Float16* s = A16 + (size_t)(rowBase + r) * lda + k0 + ks;
        const unsigned loff = (unsigned)(size_t)dst;   // LDS offset = low 32 bits
        asm volatile(
            "global_load_async_to_lds_b128 %0, %1, off\n\t"
            "global_load_async_to_lds_b128 %0, %1, off offset:16"
            :: "v"(loff), "v"(s) : "memory");
        if (kt + 1 < nkt) __builtin_prefetch(s + BK, 0, 1);
      } else {
        const float* s = A32 + (size_t)(rowBase + r) * lda + k0 + ks;
        union { _Float16 h[16]; uint4 u[2]; } t;
#pragma unroll
        for (int i = 0; i < 16; ++i) t.h[i] = (_Float16)s[i];
        ((uint4*)dst)[0] = t.u[0];
        ((uint4*)dst)[1] = t.u[1];
        if (kt + 1 < nkt) __builtin_prefetch(s + BK, 0, 1);
      }
    }
    // ---------------- stage B tile into BsT[n][k] ----------------
    if constexpr (BT) {       // global k-contiguous: straight copy per column
      const int cc = tid >> 1, ks = (tid & 1) * 16;
      _Float16* dst = &BsT[cc * LDK + ks];
      if constexpr (BH) {
        const _Float16* s = B16 + (size_t)(colBase + cc) * ldb + k0 + ks;
        const unsigned loff = (unsigned)(size_t)dst;
        asm volatile(
            "global_load_async_to_lds_b128 %0, %1, off\n\t"
            "global_load_async_to_lds_b128 %0, %1, off offset:16"
            :: "v"(loff), "v"(s) : "memory");
        if (kt + 1 < nkt) __builtin_prefetch(s + BK, 0, 1);
      } else {
        const float* s = B32 + (size_t)(colBase + cc) * ldb + k0 + ks;
        union { _Float16 h[16]; uint4 u[2]; } t;
#pragma unroll
        for (int i = 0; i < 16; ++i) t.h[i] = (_Float16)s[i];
        ((uint4*)dst)[0] = t.u[0];
        ((uint4*)dst)[1] = t.u[1];
        if (kt + 1 < nkt) __builtin_prefetch(s + BK, 0, 1);
      }
    } else {                  // global row-major KxN: 4(k)x4(n) register transpose
      const int n0l = (tid & 31) * 4, k0l = (tid >> 5) * 4;
      if constexpr (BH) {
        union { uint2 u; _Float16 h[4]; } rw[4];
#pragma unroll
        for (int kk = 0; kk < 4; ++kk)
          rw[kk].u = *(const uint2*)(B16 + (size_t)(k0 + k0l + kk) * ldb + colBase + n0l);
#pragma unroll
        for (int n = 0; n < 4; ++n) {
          union { _Float16 h[4]; uint2 u; } o;
          o.h[0] = rw[0].h[n]; o.h[1] = rw[1].h[n];
          o.h[2] = rw[2].h[n]; o.h[3] = rw[3].h[n];
          *(uint2*)&BsT[(n0l + n) * LDK + k0l] = o.u;
        }
        if (kt + 1 < nkt)
          __builtin_prefetch(B16 + (size_t)(k0 + BK + k0l) * ldb + colBase + n0l, 0, 1);
      } else {
        union { float4 v4; float f[4]; } rw[4];
#pragma unroll
        for (int kk = 0; kk < 4; ++kk)
          rw[kk].v4 = *(const float4*)(B32 + (size_t)(k0 + k0l + kk) * ldb + colBase + n0l);
#pragma unroll
        for (int n = 0; n < 4; ++n) {
          union { _Float16 h[4]; uint2 u; } o;   // k-adjacent pairs -> v_cvt_pk_f16_f32
          o.h[0] = (_Float16)rw[0].f[n]; o.h[1] = (_Float16)rw[1].f[n];
          o.h[2] = (_Float16)rw[2].f[n]; o.h[3] = (_Float16)rw[3].f[n];
          *(uint2*)&BsT[(n0l + n) * LDK + k0l] = o.u;
        }
        if (kt + 1 < nkt)
          __builtin_prefetch(B32 + (size_t)(k0 + BK + k0l) * ldb + colBase + n0l, 0, 1);
      }
    }
    if constexpr (AH || (BT && BH))
      asm volatile("s_wait_asynccnt 0" ::: "memory");
    __syncthreads();

    // ---------------- fragments (2x ds_load_b128 each) ----------------
    auto loadA = [&](int m0) -> v16h {
      union { v16h v; unsigned u[8]; } r;
      const _Float16* rp = &As[(m0 + (lane & 15)) * LDK];
      const int kb = (lane >> 4) << 4;  // lanes 16-31: K+16
#pragma unroll
      for (int i = 0; i < 8; ++i) {
        const int kk = 2 * i + kb;
        r.u[i] = *(const unsigned*)(rp + kk);
      }
      return r.v;
    };
    auto loadB = [&](int n0) -> v16h {
      union { v16h v; unsigned u[8]; } r;
      const _Float16* rp = &BsT[(n0 + (lane & 15)) * LDK];
      const int kh = (lane >> 4) << 4;  // lanes 16-31: K+16
#pragma unroll
      for (int i = 0; i < 8; ++i) r.u[i] = *(const unsigned*)(rp + kh + 2 * i);
      return r.v;
    };

    v16h af[4], bf[2];
#pragma unroll
    for (int tm = 0; tm < 4; ++tm) af[tm] = loadA(wm * 64 + tm * 16);
#pragma unroll
    for (int tn = 0; tn < 2; ++tn) bf[tn] = loadB(wn * 32 + tn * 16);
#pragma unroll
    for (int tm = 0; tm < 4; ++tm)
#pragma unroll
      for (int tn = 0; tn < 2; ++tn)
        acc[tm][tn] = __builtin_amdgcn_wmma_f32_16x16x32_f16(
            false, af[tm], false, bf[tn], (short)0, acc[tm][tn], false, false);
    __syncthreads();
  }

  // ---- epilogue: C layout M = r + 8*(lane>>4), N = lane&15 ----
#pragma unroll
  for (int tm = 0; tm < 4; ++tm)
#pragma unroll
    for (int tn = 0; tn < 2; ++tn) {
      const int gr0 = rowBase + wm * 64 + tm * 16 + ((lane >> 4) << 3);
      const int gc = colBase + wn * 32 + tn * 16 + (lane & 15);
#pragma unroll
      for (int r = 0; r < 8; ++r)
        epi_store<MODE>(p, z, gr0 + r, gc, acc[tm][tn][r]);
    }
}

// ---------------------------------------------------------------------------
// Host launcher
// ---------------------------------------------------------------------------
extern "C" void kernel_launch(void* const* d_in, const int* in_sizes, int n_in,
                              void* d_out, int out_size, void* d_ws, size_t ws_size,
                              hipStream_t stream) {
  (void)in_sizes; (void)n_in; (void)out_size; (void)ws_size;
  const float* x    = (const float*)d_in[0];
  const float* adj  = (const float*)d_in[1];
  const float* mask = (const float*)d_in[2];
  const float* dist = (const float*)d_in[3];
  const float* W1   = (const float*)d_in[4];
  const float* b1   = (const float*)d_in[5];
  const float* W2   = (const float*)d_in[6];
  const float* b2   = (const float*)d_in[7];
  const float* W3   = (const float*)d_in[8];
  const float* b3   = (const float*)d_in[9];
  const float* ln1g = (const float*)d_in[10];
  const float* ln1b = (const float*)d_in[11];
  const float* Wq   = (const float*)d_in[12];
  const float* Wk   = (const float*)d_in[13];
  const float* Wv   = (const float*)d_in[14];
  const float* Wo   = (const float*)d_in[15];
  const float* bo   = (const float*)d_in[16];
  const float* ln2g = (const float*)d_in[17];
  const float* ln2b = (const float*)d_in[18];
  const float* Wf1  = (const float*)d_in[19];
  const float* bf1  = (const float*)d_in[20];
  const float* Wf2  = (const float*)d_in[21];
  const float* bf2  = (const float*)d_in[22];

  const size_t BN_   = (size_t)Bq * Nq;                // 16384
  const size_t BNE4  = BN_ * Eq * 4;                   // 16 MB
  const size_t BHNE2 = (size_t)Bq * Hq * Nq * Eq * 2;  // 64 MB
  const size_t BHNN4 = (size_t)Bq * Hq * Nq * Nq * 4;  // 128 MB
  const size_t BHNN2 = BHNN4 / 2;                      // 64 MB

  char* w = (char*)d_ws;
  size_t o = 0;
  float* An   = (float*)(w + o); o += (size_t)Bq * Nq * Nq * 4;
  float* dis  = (float*)(w + o); o += BN_ * 4;
  float* tmpF = (float*)(w + o);                       // also FFN f16 intermediate
  _Float16* t1 = (_Float16*)tmpF;
  o += (size_t)Bq * Nq * Fq * 2;                       // 32 MB >= BNE4
  float* gcn  = (float*)(w + o); o += BNE4;
  float* gbuf = (float*)(w + o); o += BNE4;
  float* hbuf = (float*)(w + o); o += BNE4;            // also second LN output
  _Float16* qb = (_Float16*)(w + o); o += BHNE2;       // reused as y after scores
  _Float16* kb = (_Float16*)(w + o); o += BHNE2;
  _Float16* vb = (_Float16*)(w + o); o += BHNE2;       // stored transposed (B,H,E,N)
  float* Sb   = (float*)(w + o); o += BHNN4;
  _Float16* Pb = (_Float16*)(w + o); o += BHNN2;
  float* att  = (float*)(w + o); o += BNE4;
  _Float16* yb = qb;
  float* flb = hbuf;

  const dim3 blk(256);

  // 1) adjacency normalization
  k_rowdis<<<dim3((unsigned)(BN_ / 8)), blk, 0, stream>>>(adj, dis);
  k_An<<<dim3((unsigned)(BN_ * Nq / 256)), blk, 0, stream>>>(adj, dis, An);

  EpiP p;

  // 2) GCN layer 1
  p = EpiP{}; p.out_f = tmpF; p.ldc = Eq;
  gemm_wmma_k<0, false, false, false><<<dim3(Eq / 128, BN_ / 128, 1), blk, 0, stream>>>(
      x, W1, INq, INq, Eq, 0, 0, p);
  p = EpiP{}; p.out_f = gbuf; p.bias = b1; p.mask = mask; p.accum = gcn; p.first = 1;
  gemm_wmma_k<1, false, false, false><<<dim3(2, 2, Bq), blk, 0, stream>>>(
      An, tmpF, Nq, Nq, Eq, (long)Nq * Nq, (long)Nq * Eq, p);

  // 3) GCN layer 2
  p = EpiP{}; p.out_f = tmpF; p.ldc = Eq;
  gemm_wmma_k<0, false, false, false><<<dim3(Eq / 128, BN_ / 128, 1), blk, 0, stream>>>(
      gbuf, W2, Eq, Eq, Eq, 0, 0, p);
  p = EpiP{}; p.out_f = gbuf; p.bias = b2; p.mask = mask; p.accum = gcn; p.first = 0;
  gemm_wmma_k<1, false, false, false><<<dim3(2, 2, Bq), blk, 0, stream>>>(
      An, tmpF, Nq, Nq, Eq, (long)Nq * Nq, (long)Nq * Eq, p);

  // 4) GCN layer 3
  p = EpiP{}; p.out_f = tmpF; p.ldc = Eq;
  gemm_wmma_k<0, false, false, false><<<dim3(Eq / 128, BN_ / 128, 1), blk, 0, stream>>>(
      gbuf, W3, Eq, Eq, Eq, 0, 0, p);
  p = EpiP{}; p.out_f = gbuf; p.bias = b3; p.mask = mask; p.accum = gcn; p.first = 0;
  gemm_wmma_k<1, false, false, false><<<dim3(2, 2, Bq), blk, 0, stream>>>(
      An, tmpF, Nq, Nq, Eq, (long)Nq * Nq, (long)Nq * Eq, p);

  // 5) layernorm 1
  k_ln<<<dim3((unsigned)(BN_ / 8)), blk, 0, stream>>>(gcn, ln1g, ln1b, hbuf);

  // 6) Q,K -> (B,H,N,E) f16; V -> (B,H,E,N) f16 (all masked, q scaled)
  p = EpiP{}; p.out_h = qb; p.mask = mask; p.scale = 0.0625f;
  gemm_wmma_k<2, false, false, false><<<dim3(Hq * Eq / 128, BN_ / 128, 1), blk, 0, stream>>>(
      hbuf, Wq, Eq, Eq, Hq * Eq, 0, 0, p);
  p.out_h = kb; p.scale = 1.f;
  gemm_wmma_k<2, false, false, false><<<dim3(Hq * Eq / 128, BN_ / 128, 1), blk, 0, stream>>>(
      hbuf, Wk, Eq, Eq, Hq * Eq, 0, 0, p);
  p = EpiP{}; p.out_h = vb; p.mask = mask;
  gemm_wmma_k<8, false, false, false><<<dim3(Hq * Eq / 128, BN_ / 128, 1), blk, 0, stream>>>(
      hbuf, Wv, Eq, Eq, Hq * Eq, 0, 0, p);

  // 7) S = q @ k^T + dist, masked to -1e9  (batched over B*H)
  p = EpiP{}; p.out_f = Sb; p.dist = dist; p.mask = mask;
  gemm_wmma_k<3, true, true, true><<<dim3(2, 2, Bq * Hq), blk, 0, stream>>>(
      qb, kb, Eq, Eq, Eq, (long)Nq * Eq, (long)Nq * Eq, p);

  // 8) softmax + re-mask -> P f16
  k_softmax<<<dim3((unsigned)(Bq * Hq * Nq / 8)), blk, 0, stream>>>(Sb, mask, Pb);

  // 9) y = P @ v : B operand is v^T (k-contiguous) -> f16 (B,N,H*E)
  p = EpiP{}; p.out_h = yb;
  gemm_wmma_k<4, true, true, true><<<dim3(2, 2, Bq * Hq), blk, 0, stream>>>(
      Pb, vb, Nq, Nq, Nq, (long)Nq * Nq, (long)Nq * Eq, p);

  // 10) att = gcn + (y @ Wo + bo)*m
  p = EpiP{}; p.out_f = att; p.bias = bo; p.mask = mask; p.add1 = gcn;
  gemm_wmma_k<5, true, false, false><<<dim3(Eq / 128, BN_ / 128, 1), blk, 0, stream>>>(
      yb, Wo, Hq * Eq, Hq * Eq, Eq, 0, 0, p);

  // 11) layernorm 2
  k_ln<<<dim3((unsigned)(BN_ / 8)), blk, 0, stream>>>(att, ln2g, ln2b, flb);

  // 12) t1 = gelu(fl @ Wf1 + bf1) -> f16
  p = EpiP{}; p.out_h = t1; p.bias = bf1;
  gemm_wmma_k<6, false, false, false><<<dim3(Fq / 128, BN_ / 128, 1), blk, 0, stream>>>(
      flb, Wf1, Eq, Eq, Fq, 0, 0, p);

  // 13) out = gcn + att + (t1 @ Wf2 + bf2)*m
  p = EpiP{}; p.out_f = (float*)d_out; p.bias = bf2; p.mask = mask;
  p.add1 = gcn; p.add2 = att;
  gemm_wmma_k<7, true, false, false><<<dim3(Eq / 128, BN_ / 128, 1), blk, 0, stream>>>(
      t1, Wf2, Fq, Fq, Eq, 0, 0, p);
}
